// QuantizedLinear_20555713479187
// MI455X (gfx1250) — matrix operand order repaired and checked by reading, not compiled, not run on hardware
//
#include <hip/hip_runtime.h>

// ---------------------------------------------------------------------------
// Quantized linear (int4 weights, fp32 activations):
//   out[M,N] = x[M,K] * W[N,K]^T + bias[N],  M=16384, N=4096, K=4096
//   w = 0.01*(nibble-8),  b = (0.01*0.047)*qb
//
// Fast path (needs 160MB workspace):
//   prep_a_bf16: x fp32 -> bf16 once (saves 32x redundant conversion)
//   prep_w_bf16: int4 -> bf16 once  (saves 128x redundant decode)
//   qlinear_wmma_async: bf16 GEMM; staging via GLOBAL_LOAD_ASYNC_TO_LDS_B128
//     (ASYNCcnt-tracked, no VGPR round trip), double-buffered LDS,
//     1 barrier/slab, 16 v_wmma_f32_16x16x32_bf16 per slab per wave.
// Fallback (small ws): fused kernel (LDS LUT weight decode, reg prefetch).
// ---------------------------------------------------------------------------

typedef __attribute__((ext_vector_type(16))) __bf16 v16bf;
typedef __attribute__((ext_vector_type(8)))  float  v8f;

#define K_DIM 4096
#define N_DIM 4096
#define BM 128
#define BN 128
#define KT 64
#define LDST 72                     // padded LDS row stride (bf16); 144B rows
#define NSLAB (K_DIM / KT)
#define ABUF_BYTES (BM * LDST * 2)  // 18432 per buffer

// two f32 -> packed bf16 pair, round-half-up: 2 adds + 1 v_perm_b32
__device__ __forceinline__ unsigned pk_bf16(float a, float b) {
  unsigned ua = __builtin_bit_cast(unsigned, a) + 0x8000u;
  unsigned ub = __builtin_bit_cast(unsigned, b) + 0x8000u;
  return __builtin_amdgcn_perm(ub, ua, 0x07060302u);
}

union Frag { unsigned u[8]; v16bf v; };

// ---------------------------- pre-pass kernels -----------------------------

__global__ __launch_bounds__(256)
void prep_a_bf16(const float* __restrict__ x, unsigned short* __restrict__ ab) {
  const size_t i = ((size_t)blockIdx.x * 256 + threadIdx.x) * 8;
  const float4* xv = (const float4*)(x + i);
  float4 f0 = xv[0], f1 = xv[1];
  uint4 d;
  d.x = pk_bf16(f0.x, f0.y);
  d.y = pk_bf16(f0.z, f0.w);
  d.z = pk_bf16(f1.x, f1.y);
  d.w = pk_bf16(f1.z, f1.w);
  *(uint4*)(ab + i) = d;
}

__global__ __launch_bounds__(256)
void prep_w_bf16(const int* __restrict__ pw, unsigned short* __restrict__ wb) {
  const size_t i = ((size_t)blockIdx.x * 256 + threadIdx.x) * 8;  // int32 idx
  const int4* pv = (const int4*)(pw + i);
  int4 q0 = pv[0], q1 = pv[1];
  int qa[8] = {q0.x, q0.y, q0.z, q0.w, q1.x, q1.y, q1.z, q1.w};
  unsigned d[8];
#pragma unroll
  for (int c = 0; c < 8; ++c) {
    const float w0 = 0.01f * (float)((qa[c] & 15) - 8);
    const float w1 = 0.01f * (float)(((qa[c] >> 4) & 15) - 8);
    d[c] = pk_bf16(w0, w1);
  }
  uint4* o = (uint4*)(wb + i * 2);
  o[0] = make_uint4(d[0], d[1], d[2], d[3]);
  o[1] = make_uint4(d[4], d[5], d[6], d[7]);
}

// ------------------------- async-staged bf16 GEMM --------------------------

__global__ __launch_bounds__(256)
void qlinear_wmma_async(const unsigned short* __restrict__ ab,  // [M,K] bf16
                        const unsigned short* __restrict__ wb,  // [N,K] bf16
                        const int* __restrict__ qb,
                        float* __restrict__ out) {
  // Layout: [A buf0 | A buf1 | B buf0 | B buf1], each BM*LDST bf16.
  __shared__ __align__(16) unsigned short SMEM[(2 * BM + 2 * BN) * LDST];

  const int t    = threadIdx.x;
  const int lane = t & 31;
  const int wave = t >> 5;
  const int half = lane >> 4;
  const int l16  = lane & 15;
  const int wm   = wave & 3;      // 4 waves tile M: 32 rows each
  const int wn   = wave >> 2;     // 2 waves tile N: 64 cols each
  const int gm0  = blockIdx.y * BM;
  const int gn0  = blockIdx.x * BN;

  // Loader: a KT slab of a 128-row tile = 128 rows x 128B; 16B chunks.
  // Thread -> chunk (t&7), base row (t>>3), 4 rows stride 32.
  const int chunk = t & 7;
  const int r0    = t >> 3;

  // LDS byte addresses for the async VDST operand. Taking the low 32 bits of
  // the flat pointer to SMEM yields the wave-relative LDS byte address
  // (local->flat cast is {aperture_hi, lds_offset}) AND forces SMEM's address
  // to escape into the asm, so the fragment ds_loads below cannot be folded.
  unsigned ldsA[4], ldsB[4];
  int      vofs[4];               // per-lane global byte offset (A and B alike)
#pragma unroll
  for (int i = 0; i < 4; ++i) {
    const int r = r0 + 32 * i;
    vofs[i] = r * (K_DIM * 2) + chunk * 16;
    ldsA[i] = (unsigned)(size_t)&SMEM[r * LDST + chunk * 8];
    ldsB[i] = (unsigned)(size_t)&SMEM[2 * BM * LDST + r * LDST + chunk * 8];
  }

  const unsigned long long abase =
      (unsigned long long)(const char*)ab + (unsigned long long)gm0 * (K_DIM * 2);
  const unsigned long long bbase =
      (unsigned long long)(const char*)wb + (unsigned long long)gn0 * (K_DIM * 2);

  auto issue_async = [&](int slab) {
    const unsigned long long sa = abase + (unsigned long long)slab * (KT * 2);
    const unsigned long long sb = bbase + (unsigned long long)slab * (KT * 2);
    const unsigned pa = (unsigned)((slab & 1) * ABUF_BYTES);
#pragma unroll
    for (int i = 0; i < 4; ++i) {
      // GVS mode: LDS dest VGPR, 32-bit per-lane offset VGPR, 64-bit SGPR base
      asm volatile("global_load_async_to_lds_b128 %0, %1, %2"
                   :: "v"(ldsA[i] + pa), "v"(vofs[i]), "s"(sa) : "memory");
      asm volatile("global_load_async_to_lds_b128 %0, %1, %2"
                   :: "v"(ldsB[i] + pa), "v"(vofs[i]), "s"(sb) : "memory");
    }
  };

  v8f acc[2][4];
#pragma unroll
  for (int mt = 0; mt < 2; ++mt)
#pragma unroll
    for (int nt = 0; nt < 4; ++nt)
#pragma unroll
      for (int v = 0; v < 8; ++v) acc[mt][nt][v] = 0.0f;

  issue_async(0);

#pragma unroll 2
  for (int i = 0; i < NSLAB; ++i) {
    asm volatile("s_wait_asynccnt 0x0" ::: "memory");  // my slab-i writes done
    __syncthreads();                                   // everyone's done
    if (i + 1 < NSLAB) issue_async(i + 1);             // fill alternate buffer

    const unsigned short* As_ = SMEM + (i & 1) * (BM * LDST);
    const unsigned short* Bs_ = SMEM + 2 * (BM * LDST) + (i & 1) * (BN * LDST);

#pragma unroll
    for (int kk = 0; kk < KT; kk += 32) {
      Frag af[2], bf[4];
#pragma unroll
      for (int mt = 0; mt < 2; ++mt) {
        const int m = wm * 32 + mt * 16 + l16;
#pragma unroll
        for (int j = 0; j < 8; ++j) {
          // ISA 16-bit A 16x32 layout: VGPR j holds K pair
          const int k = kk + (j >> 2) * 16 + half * 8 + (j & 3) * 2;
          af[mt].u[j] = *(const unsigned*)&As_[m * LDST + k];
        }
      }
#pragma unroll
      for (int nt = 0; nt < 4; ++nt) {
        const int n = wn * 64 + nt * 16 + l16;
#pragma unroll
        for (int j = 0; j < 8; ++j) {
          // same K order as A (ISA 16-bit 16x32 layout)
          const int k = kk + (j >> 2) * 16 + half * 8 + (j & 3) * 2;
          bf[nt].u[j] = *(const unsigned*)&Bs_[n * LDST + k];
        }
      }
#pragma unroll
      for (int mt = 0; mt < 2; ++mt)
#pragma unroll
        for (int nt = 0; nt < 4; ++nt)
          acc[mt][nt] = __builtin_amdgcn_wmma_f32_16x16x32_bf16(
              false, af[mt].v, false, bf[nt].v,
              (short)0, acc[mt][nt], false, false);
    }
  }

#pragma unroll
  for (int nt = 0; nt < 4; ++nt) {
    const int n = gn0 + wn * 64 + nt * 16 + l16;
    const float b = (0.01f * 0.047f) * (float)qb[n];
#pragma unroll
    for (int mt = 0; mt < 2; ++mt) {
#pragma unroll
      for (int v = 0; v < 8; ++v) {
        const int m = gm0 + wm * 32 + mt * 16 + half * 8 + v;
        out[(size_t)m * N_DIM + n] = acc[mt][nt][v] + b;
      }
    }
  }
}

// ------------------- fused fallback (small workspace) ----------------------

__global__ __launch_bounds__(256)
void qlinear_wmma_fused(const float* __restrict__ x,
                        const int*   __restrict__ pw,
                        const int*   __restrict__ qb,
                        float*       __restrict__ out) {
  __shared__ __align__(16) unsigned short As[2][BM * LDST];
  __shared__ __align__(16) unsigned short Bs[2][BN * LDST];
  __shared__ unsigned Lut[256];

  const int t    = threadIdx.x;
  const int lane = t & 31;
  const int wave = t >> 5;
  const int half = lane >> 4;
  const int l16  = lane & 15;
  const int wm   = wave & 3;
  const int wn   = wave >> 2;
  const int gm0  = blockIdx.y * BM;
  const int gn0  = blockIdx.x * BN;

  {
    const float w0 = 0.01f * (float)((t & 15) - 8);
    const float w1 = 0.01f * (float)((t >> 4) - 8);
    Lut[t] = pk_bf16(w0, w1);
  }

  const int a_c4 = t & 15;
  const int a_r0 = t >> 4;
  const int b_g  = t & 7;
  const int b_o0 = t >> 3;

  const float4* xv = (const float4*)x;
  const int4*   pv = (const int4*)pw;

  float4 ra[8];
  int4   rb[4];

  auto load_slab = [&](int slab) {
    const int kc4 = slab * (KT / 4);
    const int kg4 = slab * (KT / 8);
#pragma unroll
    for (int i = 0; i < 8; ++i)
      ra[i] = xv[(size_t)(gm0 + a_r0 + 16 * i) * (K_DIM / 4) + kc4 + a_c4];
#pragma unroll
    for (int i = 0; i < 4; ++i)
      rb[i] = pv[(size_t)(gn0 + b_o0 + 32 * i) * (K_DIM / 8) + kg4 + b_g];
  };

  auto stage = [&](int buf) {
#pragma unroll
    for (int i = 0; i < 8; ++i) {
      const int r = a_r0 + 16 * i;
      uint2 d;
      d.x = pk_bf16(ra[i].x, ra[i].y);
      d.y = pk_bf16(ra[i].z, ra[i].w);
      *(uint2*)&As[buf][r * LDST + a_c4 * 4] = d;
    }
#pragma unroll
    for (int i = 0; i < 4; ++i) {
      const int o = b_o0 + 32 * i;
      uint4 d;
      d.x = Lut[rb[i].x];
      d.y = Lut[rb[i].y];
      d.z = Lut[rb[i].z];
      d.w = Lut[rb[i].w];
      *(uint4*)&Bs[buf][o * LDST + b_g * 8] = d;
    }
  };

  v8f acc[2][4];
#pragma unroll
  for (int mt = 0; mt < 2; ++mt)
#pragma unroll
    for (int nt = 0; nt < 4; ++nt)
#pragma unroll
      for (int v = 0; v < 8; ++v) acc[mt][nt][v] = 0.0f;

  __syncthreads();
  load_slab(0);
  stage(0);
  load_slab(1);

#pragma unroll 2
  for (int i = 0; i < NSLAB; ++i) {
    __syncthreads();
    if (i + 1 < NSLAB) stage((i + 1) & 1);
    if (i + 2 < NSLAB) load_slab(i + 2);
    const unsigned short* As_ = As[i & 1];
    const unsigned short* Bs_ = Bs[i & 1];

#pragma unroll
    for (int kk = 0; kk < KT; kk += 32) {
      Frag af[2], bf[4];
#pragma unroll
      for (int mt = 0; mt < 2; ++mt) {
        const int m = wm * 32 + mt * 16 + l16;
#pragma unroll
        for (int j = 0; j < 8; ++j) {
          const int k = kk + (j >> 2) * 16 + half * 8 + (j & 3) * 2;
          af[mt].u[j] = *(const unsigned*)&As_[m * LDST + k];
        }
      }
#pragma unroll
      for (int nt = 0; nt < 4; ++nt) {
        const int n = wn * 64 + nt * 16 + l16;
#pragma unroll
        for (int j = 0; j < 8; ++j) {
          const int k = kk + (j >> 2) * 16 + half * 8 + (j & 3) * 2;
          bf[nt].u[j] = *(const unsigned*)&Bs_[n * LDST + k];
        }
      }
#pragma unroll
      for (int mt = 0; mt < 2; ++mt)
#pragma unroll
        for (int nt = 0; nt < 4; ++nt)
          acc[mt][nt] = __builtin_amdgcn_wmma_f32_16x16x32_bf16(
              false, af[mt].v, false, bf[nt].v,
              (short)0, acc[mt][nt], false, false);
    }
  }

#pragma unroll
  for (int nt = 0; nt < 4; ++nt) {
    const int n = gn0 + wn * 64 + nt * 16 + l16;
    const float b = (0.01f * 0.047f) * (float)qb[n];
#pragma unroll
    for (int mt = 0; mt < 2; ++mt) {
#pragma unroll
      for (int v = 0; v < 8; ++v) {
        const int m = gm0 + wm * 32 + mt * 16 + half * 8 + v;
        out[(size_t)m * N_DIM + n] = acc[mt][nt][v] + b;
      }
    }
  }
}

// ------------------------------- launcher ----------------------------------

extern "C" void kernel_launch(void* const* d_in, const int* in_sizes, int n_in,
                              void* d_out, int out_size, void* d_ws, size_t ws_size,
                              hipStream_t stream) {
  const float* x   = (const float*)d_in[0];
  const int*   pw  = (const int*)d_in[1];
  const int*   qb  = (const int*)d_in[2];
  float*       out = (float*)d_out;

  const int M = in_sizes[0] / K_DIM;          // 16384 for reference shapes
  const size_t aElems = (size_t)M * K_DIM;    // bf16 elements
  const size_t wElems = (size_t)N_DIM * K_DIM;
  const size_t need   = (aElems + wElems) * sizeof(unsigned short);

  dim3 grid(N_DIM / BN, M / BM);
  dim3 block(256);

  if (ws_size >= need) {
    unsigned short* abuf = (unsigned short*)d_ws;
    unsigned short* wbuf = abuf + aElems;
    prep_a_bf16<<<(unsigned)(aElems / 2048), 256, 0, stream>>>(x, abuf);
    prep_w_bf16<<<(unsigned)(wElems / 2 / 2048), 256, 0, stream>>>(pw, wbuf);
    qlinear_wmma_async<<<grid, block, 0, stream>>>(abuf, wbuf, qb, out);
  } else {
    qlinear_wmma_fused<<<grid, block, 0, stream>>>(x, pw, qb, out);
  }
}
